// SageAttnCogVideoXAttnProcessor_34325378629689
// MI455X (gfx1250) — matrix operand order repaired and checked by reading, not compiled, not run on hardware
//
#include <hip/hip_runtime.h>

#define HEADS 30
#define HD    64
#define DIM   1920
#define TXT   226
#define VID   2048
#define STOT  2274          // TXT + VID
#define SP    2288          // padded to 143*16
#define NT    143           // seq tiles of 16
#define KSTEPS 60           // DIM / 32
#define EPSV  1e-5f

typedef __attribute__((ext_vector_type(16))) _Float16 v16h;
typedef __attribute__((ext_vector_type(8)))  float    v8f;

__device__ __forceinline__ v8f wmma_f16(v16h a, v16h b, v8f c) {
  // D = A(16x32 f16) * B(32x16 f16) + C(16x16 f32)
  return __builtin_amdgcn_wmma_f32_16x16x32_f16(false, a, false, b, (short)0, c,
                                                false, false);
}

// ---------------------------------------------------------------------------
// Kernel 0: one-time f32 -> f16 conversion of the four weight matrices.
// blockIdx.y selects the matrix; each thread converts 4 contiguous elements.
// ---------------------------------------------------------------------------
__global__ __launch_bounds__(256) void cvt_w_kernel(
    const float* __restrict__ w0, const float* __restrict__ w1,
    const float* __restrict__ w2, const float* __restrict__ w3,
    _Float16* __restrict__ o0, _Float16* __restrict__ o1,
    _Float16* __restrict__ o2, _Float16* __restrict__ o3) {
  const float* src;
  _Float16* dst;
  switch (blockIdx.y) {
    case 0:  src = w0; dst = o0; break;
    case 1:  src = w1; dst = o1; break;
    case 2:  src = w2; dst = o2; break;
    default: src = w3; dst = o3; break;
  }
  const size_t idx = ((size_t)blockIdx.x * 256 + threadIdx.x) * 4;
  if (idx + 3 < (size_t)DIM * DIM) {
    float4 v = *(const float4*)(src + idx);
    dst[idx + 0] = (_Float16)v.x;
    dst[idx + 1] = (_Float16)v.y;
    dst[idx + 2] = (_Float16)v.z;
    dst[idx + 3] = (_Float16)v.w;
  }
}

// ---------------------------------------------------------------------------
// Kernel 1: fused QKV projection + bias + per-head LayerNorm + RoPE.
// One wave computes a 16(seq) x 64(one head) tile of q, k, v.
//   A lane layout (16-bit A 16x32): lane holds row = lane%16;
//     elems 0..7  -> K = ks*32 + half*16 + i
//     elems 8..15 -> K = ks*32 + half*16 + 8 + i      (half = lane/16)
//   B lane layout (32x16): lane holds col n = lane%16;
//     elems 0..15 -> K = ks*32 + half*16 + i (contiguous along W16's row n).
//   C/D layout: lane holds col n = lane%16, rows i + 8*half in v8f slot i.
// ---------------------------------------------------------------------------
__global__ __launch_bounds__(32) void proj_qkv_kernel(
    const float* __restrict__ hid, const float* __restrict__ enc,
    const float* __restrict__ cosb, const float* __restrict__ sinb,
    const _Float16* __restrict__ Wq, const float* __restrict__ bq,
    const _Float16* __restrict__ Wk, const float* __restrict__ bk,
    const _Float16* __restrict__ Wv, const float* __restrict__ bv,
    const float* __restrict__ lqw, const float* __restrict__ lqb,
    const float* __restrict__ lkw, const float* __restrict__ lkb,
    _Float16* __restrict__ Qf, _Float16* __restrict__ Kf,
    _Float16* __restrict__ Vt) {
  const int tid  = threadIdx.x;
  const int lane = tid & 15;
  const int half = tid >> 4;
  const int st   = blockIdx.x;   // seq tile
  const int h    = blockIdx.y;   // head

  // A-matrix source row for this lane (concat of encoder + hidden).
  const int sA = st * 16 + lane;
  const float* rowp = nullptr;
  if (sA < TXT)        rowp = enc + (size_t)sA * DIM;
  else if (sA < STOT)  rowp = hid + (size_t)(sA - TXT) * DIM;

  v8f zero = {};
  v8f accq[4], acck[4], accv[4];
#pragma unroll
  for (int c = 0; c < 4; ++c) { accq[c] = zero; acck[c] = zero; accv[c] = zero; }

  for (int ks = 0; ks < KSTEPS; ++ks) {
    v16h a;
    if (rowp) {
      const float* p0 = rowp + ks * 32 + half * 16;
      const float* p1 = p0 + 8;
#pragma unroll
      for (int i = 0; i < 8; ++i) {
        a[i]     = (_Float16)p0[i];
        a[8 + i] = (_Float16)p1[i];
      }
    } else {
#pragma unroll
      for (int i = 0; i < 16; ++i) a[i] = (_Float16)0.f;
    }
#pragma unroll
    for (int c = 0; c < 4; ++c) {
      const int ng = h * HD + c * 16 + lane;           // output column
      const int ko = ks * 32 + half * 16;
      const _Float16* pq = Wq + (size_t)ng * DIM + ko;
      const _Float16* pk = Wk + (size_t)ng * DIM + ko;
      const _Float16* pv = Wv + (size_t)ng * DIM + ko;
      v16h bqv, bkv, bvv;
#pragma unroll
      for (int i = 0; i < 16; ++i) {
        bqv[i] = pq[i];
        bkv[i] = pk[i];
        bvv[i] = pv[i];
      }
      accq[c] = wmma_f16(a, bqv, accq[c]);
      acck[c] = wmma_f16(a, bkv, acck[c]);
      accv[c] = wmma_f16(a, bvv, accv[c]);
    }
  }

  // Bias add (same per column across rows).
#pragma unroll
  for (int c = 0; c < 4; ++c) {
    const int ng = h * HD + c * 16 + lane;
    const float bqs = bq[ng], bks = bk[ng], bvs = bv[ng];
#pragma unroll
    for (int i = 0; i < 8; ++i) {
      accq[c][i] += bqs; acck[c][i] += bks; accv[c][i] += bvs;
    }
  }

  // Per-row (64-dim) LayerNorm for q and k.
  float wq_[4], bq_[4], wk_[4], bk_[4];
#pragma unroll
  for (int c = 0; c < 4; ++c) {
    const int d = c * 16 + lane;
    wq_[c] = lqw[d]; bq_[c] = lqb[d];
    wk_[c] = lkw[d]; bk_[c] = lkb[d];
  }
#pragma unroll
  for (int i = 0; i < 8; ++i) {
    float s1 = 0.f, s2 = 0.f;
#pragma unroll
    for (int c = 0; c < 4; ++c) { float x = accq[c][i]; s1 += x; s2 += x * x; }
#pragma unroll
    for (int off = 1; off < 16; off <<= 1) {
      s1 += __shfl_xor(s1, off, 32);
      s2 += __shfl_xor(s2, off, 32);
    }
    float mu = s1 * (1.f / 64.f);
    float rs = rsqrtf(s2 * (1.f / 64.f) - mu * mu + EPSV);
#pragma unroll
    for (int c = 0; c < 4; ++c)
      accq[c][i] = (accq[c][i] - mu) * rs * wq_[c] + bq_[c];

    s1 = 0.f; s2 = 0.f;
#pragma unroll
    for (int c = 0; c < 4; ++c) { float x = acck[c][i]; s1 += x; s2 += x * x; }
#pragma unroll
    for (int off = 1; off < 16; off <<= 1) {
      s1 += __shfl_xor(s1, off, 32);
      s2 += __shfl_xor(s2, off, 32);
    }
    mu = s1 * (1.f / 64.f);
    rs = rsqrtf(s2 * (1.f / 64.f) - mu * mu + EPSV);
#pragma unroll
    for (int c = 0; c < 4; ++c)
      acck[c][i] = (acck[c][i] - mu) * rs * wk_[c] + bk_[c];
  }

  // RoPE on video rows (rows of the C-layout: rg = st*16 + i + 8*half).
#pragma unroll
  for (int i = 0; i < 8; ++i) {
    const int rg = st * 16 + i + 8 * half;
    const bool isvid = (rg >= TXT) && (rg < STOT);
    const int vr = rg - TXT;
#pragma unroll
    for (int c = 0; c < 4; ++c) {
      const int d = c * 16 + lane;
      const float xq = accq[c][i], xk = acck[c][i];
      const float pq = __shfl_xor(xq, 1, 32);   // value at dim d^1
      const float pk = __shfl_xor(xk, 1, 32);
      const float rq = (d & 1) ? pq : -pq;
      const float rk = (d & 1) ? pk : -pk;
      float cv = 1.f, sv = 0.f;
      if (isvid) {
        cv = cosb[(size_t)vr * HD + d];
        sv = sinb[(size_t)vr * HD + d];
      }
      accq[c][i] = xq * cv + rq * sv;
      acck[c][i] = xk * cv + rk * sv;
    }
  }

  // Store: Q,K row-major f16 [head][s][64]; V transposed f16 [head][64][s].
#pragma unroll
  for (int i = 0; i < 8; ++i) {
    const int rg = st * 16 + i + 8 * half;
    const bool ok = rg < STOT;
#pragma unroll
    for (int c = 0; c < 4; ++c) {
      const int d = c * 16 + lane;
      Qf[((size_t)h * SP + rg) * HD + d] = (_Float16)(ok ? accq[c][i] : 0.f);
      Kf[((size_t)h * SP + rg) * HD + d] = (_Float16)(ok ? acck[c][i] : 0.f);
      Vt[((size_t)h * HD + d) * SP + rg] = (_Float16)(ok ? accv[c][i] : 0.f);
    }
  }
}

// ---------------------------------------------------------------------------
// Kernel 2: flash attention, one wave per (16-query tile, head).
// ---------------------------------------------------------------------------
__global__ __launch_bounds__(32) void attn_kernel(
    const _Float16* __restrict__ Qf, const _Float16* __restrict__ Kf,
    const _Float16* __restrict__ Vt, _Float16* __restrict__ attn) {
  __shared__ _Float16 ps[16 * 17];   // P tile staging (C-layout -> A-layout)
  const int tid  = threadIdx.x;
  const int lane = tid & 15;
  const int half = tid >> 4;
  const int qt   = blockIdx.x;
  const int h    = blockIdx.y;

  // Preload Q A-tiles for K-dim steps 0..31 and 32..63.
  const _Float16* qp = Qf + ((size_t)h * SP + qt * 16 + lane) * HD;
  v16h a0, a1;
#pragma unroll
  for (int i = 0; i < 8; ++i) {
    a0[i]     = qp[half * 16 + i];
    a0[8 + i] = qp[half * 16 + 8 + i];
    a1[i]     = qp[32 + half * 16 + i];
    a1[8 + i] = qp[40 + half * 16 + i];
  }

  float m[8], l[8];
  v8f zero = {};
  v8f o[4];
#pragma unroll
  for (int i = 0; i < 8; ++i) { m[i] = -3.0e38f; l[i] = 0.f; }
#pragma unroll
  for (int c = 0; c < 4; ++c) o[c] = zero;

  const float scale = 0.125f;                  // 1/sqrt(64)
  const int key0 = lane;                       // this lane's key column in tile

  for (int j = 0; j < NT; ++j) {
    // S = Q * K^T : B[k][n] = K[key n][dim k] -> contiguous in K's row.
    const _Float16* kp = Kf + ((size_t)h * SP + j * 16 + lane) * HD;
    v16h b0, b1;
#pragma unroll
    for (int i = 0; i < 16; ++i) {
      b0[i] = kp[half * 16 + i];
      b1[i] = kp[32 + half * 16 + i];
    }
    v8f sv = zero;
    sv = wmma_f16(a0, b0, sv);
    sv = wmma_f16(a1, b1, sv);

    const int key = j * 16 + key0;
    float sc[8];
#pragma unroll
    for (int i = 0; i < 8; ++i)
      sc[i] = (key < STOT) ? sv[i] * scale : -1.0e30f;

    float p[8];
#pragma unroll
    for (int i = 0; i < 8; ++i) {
      float t = sc[i];
#pragma unroll
      for (int off = 1; off < 16; off <<= 1) t = fmaxf(t, __shfl_xor(t, off, 32));
      const float mn = fmaxf(m[i], t);
      const float corr = __expf(m[i] - mn);
      m[i] = mn;
      const float pe = __expf(sc[i] - mn);
      float psum = pe;
#pragma unroll
      for (int off = 1; off < 16; off <<= 1) psum += __shfl_xor(psum, off, 32);
      l[i] = l[i] * corr + psum;
#pragma unroll
      for (int c = 0; c < 4; ++c) o[c][i] *= corr;
      p[i] = pe;
    }

    // C-layout -> A-layout transpose of P via LDS (single wave: DS in-order).
#pragma unroll
    for (int i = 0; i < 8; ++i)
      ps[(i + 8 * half) * 17 + lane] = (_Float16)p[i];
    asm volatile("s_wait_dscnt 0" ::: "memory");

    v16h pa;
#pragma unroll
    for (int i = 0; i < 8; ++i) {
      pa[i]     = half ? (_Float16)0.f : ps[lane * 17 + i];      // K = 16*half+i (half 0 valid)
      pa[8 + i] = half ? (_Float16)0.f : ps[lane * 17 + 8 + i];  // K = 16..31 zero-padded
    }

    // O += P * V : B[k][n] = V[key k][dim n] = Vt[dim][key] contiguous.
#pragma unroll
    for (int c = 0; c < 4; ++c) {
      const _Float16* vp =
          Vt + ((size_t)h * HD + c * 16 + lane) * SP + j * 16;
      v16h vb;
#pragma unroll
      for (int i = 0; i < 16; ++i) vb[i] = vp[i];
      o[c] = wmma_f16(pa, vb, o[c]);
    }
  }

  // Normalize and store to attn[s][h*64+d] (f16, all padded rows finite).
#pragma unroll
  for (int i = 0; i < 8; ++i) {
    const float inv = 1.f / l[i];
    const int rg = qt * 16 + i + 8 * half;
#pragma unroll
    for (int c = 0; c < 4; ++c)
      attn[(size_t)rg * DIM + h * HD + c * 16 + lane] =
          (_Float16)(o[c][i] * inv);
  }
}

// ---------------------------------------------------------------------------
// Kernel 3: output projection + bias + (video-first, text-second) reorder.
// ---------------------------------------------------------------------------
__global__ __launch_bounds__(32) void oproj_kernel(
    const _Float16* __restrict__ attn, const _Float16* __restrict__ Wo,
    const float* __restrict__ bo, float* __restrict__ out) {
  const int tid  = threadIdx.x;
  const int lane = tid & 15;
  const int half = tid >> 4;
  const int st   = blockIdx.x;   // seq tile
  const int nt   = blockIdx.y;   // 64-wide output column tile

  const _Float16* ap = attn + (size_t)(st * 16 + lane) * DIM;
  v8f zero = {};
  v8f acc[4];
#pragma unroll
  for (int c = 0; c < 4; ++c) acc[c] = zero;

  for (int ks = 0; ks < KSTEPS; ++ks) {
    v16h a;
#pragma unroll
    for (int i = 0; i < 8; ++i) {
      a[i]     = ap[ks * 32 + half * 16 + i];
      a[8 + i] = ap[ks * 32 + half * 16 + 8 + i];
    }
#pragma unroll
    for (int c = 0; c < 4; ++c) {
      const int ng = nt * 64 + c * 16 + lane;
      const _Float16* wp = Wo + (size_t)ng * DIM + ks * 32 + half * 16;
      __builtin_prefetch(wp + 32, 0, 0);
      v16h b;
#pragma unroll
      for (int i = 0; i < 16; ++i) b[i] = wp[i];
      acc[c] = wmma_f16(a, b, acc[c]);
    }
  }

#pragma unroll
  for (int i = 0; i < 8; ++i) {
    const int rg = st * 16 + i + 8 * half;
    if (rg < STOT) {
      const int r = (rg >= TXT) ? (rg - TXT) : (VID + rg);
#pragma unroll
      for (int c = 0; c < 4; ++c) {
        const int ng = nt * 64 + c * 16 + lane;
        out[(size_t)r * DIM + ng] = acc[c][i] + bo[ng];
      }
    }
  }
}

// ---------------------------------------------------------------------------
extern "C" void kernel_launch(void* const* d_in, const int* in_sizes, int n_in,
                              void* d_out, int out_size, void* d_ws,
                              size_t ws_size, hipStream_t stream) {
  const float* hid = (const float*)d_in[0];
  const float* enc = (const float*)d_in[1];
  const float* rc  = (const float*)d_in[2];
  const float* rs  = (const float*)d_in[3];
  const float* Wq  = (const float*)d_in[4];
  const float* bq  = (const float*)d_in[5];
  const float* Wk  = (const float*)d_in[6];
  const float* bk  = (const float*)d_in[7];
  const float* Wv  = (const float*)d_in[8];
  const float* bv  = (const float*)d_in[9];
  const float* lqw = (const float*)d_in[10];
  const float* lqb = (const float*)d_in[11];
  const float* lkw = (const float*)d_in[12];
  const float* lkb = (const float*)d_in[13];
  const float* Wo  = (const float*)d_in[14];
  const float* bo  = (const float*)d_in[15];

  const size_t NELT = (size_t)HEADS * SP * HD;   // 4,392,960 (== SP*DIM)
  const size_t WELT = (size_t)DIM * DIM;         // 3,686,400
  _Float16* Qf   = (_Float16*)d_ws;
  _Float16* Kf   = Qf + NELT;
  _Float16* Vt   = Kf + NELT;
  _Float16* attn = Vt + NELT;
  _Float16* Wq16 = attn + NELT;
  _Float16* Wk16 = Wq16 + WELT;
  _Float16* Wv16 = Wk16 + WELT;
  _Float16* Wo16 = Wv16 + WELT;

  // One-time weight conversion: 4 matrices, 4 elems/thread, 256 thr/block.
  const int cvt_blocks = (int)(WELT / (256 * 4));   // 3600, exact
  cvt_w_kernel<<<dim3(cvt_blocks, 4), 256, 0, stream>>>(
      Wq, Wk, Wv, Wo, Wq16, Wk16, Wv16, Wo16);

  proj_qkv_kernel<<<dim3(NT, HEADS), 32, 0, stream>>>(
      hid, enc, rc, rs, Wq16, bq, Wk16, bk, Wv16, bv, lqw, lqb, lkw, lkb, Qf,
      Kf, Vt);
  attn_kernel<<<dim3(NT, HEADS), 32, 0, stream>>>(Qf, Kf, Vt, attn);
  oproj_kernel<<<dim3(NT, HEADS), 32, 0, stream>>>(attn, Wo16, bo,
                                                   (float*)d_out);
}
